// MultiHeadAttention_28750511079639
// MI455X (gfx1250) — matrix operand order repaired and checked by reading, not compiled, not run on hardware
//
#include <hip/hip_runtime.h>
#include <hip/hip_bf16.h>

// ---------------- problem constants ----------------
constexpr int cB   = 2;
constexpr int cT   = 2048;
constexpr int cH   = 2048;
constexpr int cNH  = 32;
constexpr int cNKV = 8;
constexpr int cHD  = 64;   // head dim
// ---------------------------------------------------

typedef __attribute__((ext_vector_type(16))) __bf16 v16bf;
typedef __attribute__((ext_vector_type(8)))  __bf16 v8bf;
typedef __attribute__((ext_vector_type(8)))  float  v8f;

// ---- WMMA wrapper: D = A(16x32 bf16) * B(32x16 bf16) + C(16x16 f32) ----
__device__ inline v8f wmma_bf16(v16bf a, v16bf b, v8f c) {
    return __builtin_amdgcn_wmma_f32_16x16x32_bf16(
        /*neg_a=*/false, a, /*neg_b=*/false, b,
        /*c_mod=*/(short)0, c, /*reuse_a=*/false, /*reuse_b=*/false);
}

// A fragment (16 rows x 32 K), row-major source, leading dim ld (elements).
// ISA: lane L holds row M=L%16; lanes 0-15 K {0..7,16..23}, lanes 16-31 K {8..15,24..31}.
__device__ inline v16bf frag_a(const __bf16* src, int ld, int lane) {
    const int r  = lane & 15;
    const int hf = lane >> 4;
    const __bf16* p = src + (size_t)r * ld + hf * 8;
    v8bf lo = *(const v8bf*)p;
    v8bf hi = *(const v8bf*)(p + 16);
    return __builtin_shufflevector(lo, hi, 0,1,2,3,4,5,6,7,8,9,10,11,12,13,14,15);
}

// B fragment (32 K x 16 cols), source row-major [col][K]:
// lane L holds column N=L%16; lanes 0-15 K {0..7,16..23}, lanes 16-31 K {8..15,24..31}.
__device__ inline v16bf frag_b(const __bf16* src, int ld, int lane) {
    const __bf16* p = src + (size_t)(lane & 15) * ld + (lane >> 4) * 8;
    v8bf lo = *(const v8bf*)p;
    v8bf hi = *(const v8bf*)(p + 16);
    return __builtin_shufflevector(lo, hi, 0,1,2,3,4,5,6,7,8,9,10,11,12,13,14,15);
}

// ---- CDNA5 async global->LDS copy (16B/lane), tracked by ASYNCcnt ----
__device__ inline void async_copy_b128(unsigned lds_addr, const __bf16* gsrc) {
    asm volatile("global_load_async_to_lds_b128 %0, %1, off"
                 :: "v"(lds_addr), "v"((unsigned long long)(size_t)gsrc)
                 : "memory");
}
template <int N>
__device__ inline void wait_async() {
    asm volatile("s_wait_asynccnt %0" :: "i"(N) : "memory");
}

// ---------------- fp32 -> bf16 convert ----------------
__global__ void mha_cvt_bf16(const float* __restrict__ in, __bf16* __restrict__ out, int n) {
    int i = blockIdx.x * blockDim.x + threadIdx.x;
    int stride = gridDim.x * blockDim.x;
    for (; i < n; i += stride) out[i] = (__bf16)in[i];
}

// ---------------- GEMM: C[M,N] = A[M,K] @ W[N,K]^T ----------------
// MODE 0: bf16 out, row-major [M,N]
// MODE 1: bf16 out, KV layout: (m->(b,t), n->(kv,d)) => [(b*NKV+kv)*T + t]*HD + d
// MODE 2: f32 out, row-major [M,N]
// MI: number of 16-row M fragments per wave (wave tile = MI*16 x 64)
template <int MODE, int MI>
__global__ __launch_bounds__(128)
void mha_gemm_bf16(const __bf16* __restrict__ A, const __bf16* __restrict__ W,
                   __bf16* __restrict__ Cb, float* __restrict__ Cf,
                   int M, int N, int K) {
    const int lane = threadIdx.x & 31;
    const int wave = threadIdx.x >> 5;
    const int n0 = blockIdx.x * 64;
    const int m0 = blockIdx.y * (64 * MI) + wave * (16 * MI);

    const __bf16* Arow = A + (size_t)m0 * K;
    v8f c[MI][4];
#pragma unroll
    for (int i = 0; i < MI; ++i)
#pragma unroll
        for (int j = 0; j < 4; ++j) c[i][j] = (v8f){};

    for (int k0 = 0; k0 < K; k0 += 32) {
        v16bf a[MI];
#pragma unroll
        for (int i = 0; i < MI; ++i)
            a[i] = frag_a(Arow + (size_t)(i * 16) * K + k0, K, lane);
        if (k0 + 128 < K)   // prefetch A stream a few tiles ahead
            __builtin_prefetch(Arow + (size_t)(lane & 15) * K + k0 + 128, 0, 3);
#pragma unroll
        for (int j = 0; j < 4; ++j) {
            v16bf b = frag_b(W + (size_t)(n0 + j * 16) * K + k0, K, lane);
#pragma unroll
            for (int i = 0; i < MI; ++i)
                c[i][j] = wmma_bf16(a[i], b, c[i][j]);
        }
    }

    const int half = lane >> 4;
    const int col  = lane & 15;
#pragma unroll
    for (int i = 0; i < MI; ++i) {
#pragma unroll
        for (int j = 0; j < 4; ++j) {
#pragma unroll
            for (int r = 0; r < 8; ++r) {
                const int m = m0 + i * 16 + r + half * 8;   // C layout: VGPR r -> M=r (+8 upper lanes)
                const int n = n0 + j * 16 + col;
                const float v = c[i][j][r];
                if (MODE == 0) {
                    Cb[(size_t)m * N + n] = (__bf16)v;
                } else if (MODE == 1) {
                    const int bb = m / cT, t = m % cT;
                    const int kv = n >> 6, d = n & 63;
                    Cb[(((size_t)bb * cNKV + kv) * cT + t) * cHD + d] = (__bf16)v;
                } else {
                    Cf[(size_t)m * N + n] = v;
                }
            }
        }
    }
}

// ---------------- RoPE (in place, bf16) ----------------
// Pair j rotates dims (2j, 2j+1); angle(d) = t * BASE^(-(d%32)/32)
__global__ void mha_rope(__bf16* __restrict__ buf, int npairs, int tdiv) {
    const int idx = blockIdx.x * blockDim.x + threadIdx.x;
    if (idx >= npairs) return;
    const int j = idx & 31;
    const int t = (idx / tdiv) % cT;
    const float x0 = (float)buf[2 * idx];
    const float x1 = (float)buf[2 * idx + 1];
    const int i0 = (2 * j) & 31;
    const int i1 = (2 * j + 1) & 31;
    const float f0 = __powf(10000.0f, -(float)i0 / 32.0f);
    const float f1 = __powf(10000.0f, -(float)i1 / 32.0f);
    const float a0 = (float)t * f0;
    const float a1 = (float)t * f1;
    buf[2 * idx]     = (__bf16)(x0 * __cosf(a0) - x1 * __sinf(a0));
    buf[2 * idx + 1] = (__bf16)(x1 * __cosf(a1) + x0 * __sinf(a1));
}

// ---------------- Flash attention (double-buffered async staging) ----------------
// grid (T/64, NH, B), 128 threads = 4 waves, each wave = 16 query rows.
// Q layout [B,T,NH*HD]; K,V layout [B,NKV,T,HD]; O layout [B,T,NH*HD] (bf16).
// K/V tiles streamed with CDNA5 async global->LDS copies; tile i+1 DMA overlaps
// tile i compute via s_wait_asynccnt 4 (async loads retire in order).
__global__ __launch_bounds__(128)
void mha_attn(const __bf16* __restrict__ Q, const __bf16* __restrict__ K,
              const __bf16* __restrict__ V, __bf16* __restrict__ O) {
    __shared__ __bf16 kt[2][32 * 64];      // [buf][key][d]
    __shared__ __bf16 vraw[2][32 * 64];    // [buf][key][d] (async landing zone)
    __shared__ __bf16 vt[2][64 * 32];      // [buf][d][key] (transposed V)
    __shared__ __bf16 pbuf[4][16 * 32];    // per-wave P tile [row][key]

    const int tid  = threadIdx.x;
    const int lane = tid & 31;
    const int wave = tid >> 5;
    const int half = lane >> 4;
    const int col  = lane & 15;

    const int qb = blockIdx.x * 64;
    const int h  = blockIdx.y;
    const int b  = blockIdx.z;
    const int kv = h / (cNH / cNKV);
    const int qrow0 = qb + wave * 16;

    const __bf16* Qbase = Q + ((size_t)b * cT + qrow0) * cH + h * cHD;
    const v16bf qf0 = frag_a(Qbase, cH, lane);        // d = 0..31
    const v16bf qf1 = frag_a(Qbase + 32, cH, lane);   // d = 32..63

    const __bf16* Kh = K + ((size_t)b * cNKV + kv) * (size_t)cT * cHD;
    const __bf16* Vh = V + ((size_t)b * cNKV + kv) * (size_t)cT * cHD;

    const unsigned kt_base   = (unsigned)(size_t)(&kt[0][0]);
    const unsigned vraw_base = (unsigned)(size_t)(&vraw[0][0]);

    // my two 16B chunks of a 32x64 tile: chunk -> (key, d0)
    const int key_c0 = tid >> 2,            d0_c0 = (tid & 3) * 16;
    const int key_c1 = (tid + 128) >> 2,    d0_c1 = ((tid + 128) & 3) * 16;
    const unsigned loff_c0 = (unsigned)((key_c0 * 64 + d0_c0) * 2);
    const unsigned loff_c1 = (unsigned)((key_c1 * 64 + d0_c1) * 2);

    auto issue_stage = [&](int buf, int k0) {
        const unsigned bo = (unsigned)buf * (32 * 64 * 2);
        async_copy_b128(kt_base + bo + loff_c0,   Kh + (size_t)(k0 + key_c0) * cHD + d0_c0);
        async_copy_b128(vraw_base + bo + loff_c0, Vh + (size_t)(k0 + key_c0) * cHD + d0_c0);
        async_copy_b128(kt_base + bo + loff_c1,   Kh + (size_t)(k0 + key_c1) * cHD + d0_c1);
        async_copy_b128(vraw_base + bo + loff_c1, Vh + (size_t)(k0 + key_c1) * cHD + d0_c1);
    };

    v8f o0 = {}, o1 = {}, o2 = {}, o3 = {};
    float m_[8], l_[8];
#pragma unroll
    for (int r = 0; r < 8; ++r) { m_[r] = -1e30f; l_[r] = 0.0f; }

    const float scale = 0.125f;             // 1/sqrt(64)
    const int nsteps = (qb + 64) >> 5;      // causal: keys < qb+64

    issue_stage(0, 0);                      // prime the pipeline

    for (int i = 0; i < nsteps; ++i) {
        const int cur = i & 1;
        const int k0  = i * 32;

        if (i + 1 < nsteps) {
            // buf[1-cur] was last read in iteration i-1; end-of-iter barrier passed.
            issue_stage(1 - cur, k0 + 32);
            wait_async<4>();                // oldest 4 (tile i) resident; newest 4 stream on
        } else {
            wait_async<0>();
        }

        // transpose my own vraw chunks into vt[cur] (thread-local async dependency)
        {
            v8bf x0 = *(const v8bf*)&vraw[cur][key_c0 * 64 + d0_c0];
            v8bf x1 = *(const v8bf*)&vraw[cur][key_c0 * 64 + d0_c0 + 8];
            v8bf y0 = *(const v8bf*)&vraw[cur][key_c1 * 64 + d0_c1];
            v8bf y1 = *(const v8bf*)&vraw[cur][key_c1 * 64 + d0_c1 + 8];
#pragma unroll
            for (int e = 0; e < 8; ++e) {
                vt[cur][(d0_c0 + e) * 32 + key_c0]     = x0[e];
                vt[cur][(d0_c0 + 8 + e) * 32 + key_c0] = x1[e];
                vt[cur][(d0_c1 + e) * 32 + key_c1]     = y0[e];
                vt[cur][(d0_c1 + 8 + e) * 32 + key_c1] = y1[e];
            }
        }
        __syncthreads();   // all waves' kt[cur] (async) + vt[cur] (ds) visible

        const __bf16* ktc = &kt[cur][0];
        const __bf16* vtc = &vt[cur][0];

        // --- S = Q K^T : two 16x16 tiles (keys k0..k0+15, k0+16..k0+31) ---
        v8f s0 = {}, s1 = {};
        s0 = wmma_bf16(qf0, frag_b(ktc, 64, lane), s0);
        s0 = wmma_bf16(qf1, frag_b(ktc + 32, 64, lane), s0);
        s1 = wmma_bf16(qf0, frag_b(ktc + 16 * 64, 64, lane), s1);
        s1 = wmma_bf16(qf1, frag_b(ktc + 16 * 64 + 32, 64, lane), s1);

        const int key0 = k0 + col;
        const int key1 = k0 + 16 + col;
        float rowmax[8];
#pragma unroll
        for (int r = 0; r < 8; ++r) {
            const int q = qrow0 + r + half * 8;
            const float a = (key0 <= q) ? s0[r] * scale : -1e30f;
            const float bq = (key1 <= q) ? s1[r] * scale : -1e30f;
            s0[r] = a; s1[r] = bq;
            float t = fmaxf(a, bq);
            t = fmaxf(t, __shfl_xor(t, 1, 32));
            t = fmaxf(t, __shfl_xor(t, 2, 32));
            t = fmaxf(t, __shfl_xor(t, 4, 32));
            t = fmaxf(t, __shfl_xor(t, 8, 32));
            rowmax[r] = t;
        }
#pragma unroll
        for (int r = 0; r < 8; ++r) {
            const float mn = fmaxf(m_[r], rowmax[r]);
            const float alpha = __expf(m_[r] - mn);
            const float p0 = __expf(s0[r] - mn);
            const float p1 = __expf(s1[r] - mn);
            float s = p0 + p1;
            s += __shfl_xor(s, 1, 32);
            s += __shfl_xor(s, 2, 32);
            s += __shfl_xor(s, 4, 32);
            s += __shfl_xor(s, 8, 32);
            l_[r] = l_[r] * alpha + s;
            m_[r] = mn;
            o0[r] *= alpha; o1[r] *= alpha; o2[r] *= alpha; o3[r] *= alpha;
            const int row = r + half * 8;
            pbuf[wave][row * 32 + col]      = (__bf16)p0;
            pbuf[wave][row * 32 + 16 + col] = (__bf16)p1;
        }

        // --- O += P @ V  (P: 16x32 A-frag from LDS; V^T rows give B-frags) ---
        const v16bf pf = frag_a(&pbuf[wave][0], 32, lane);
        o0 = wmma_bf16(pf, frag_b(vtc + 0 * 16 * 32, 32, lane), o0);
        o1 = wmma_bf16(pf, frag_b(vtc + 1 * 16 * 32, 32, lane), o1);
        o2 = wmma_bf16(pf, frag_b(vtc + 2 * 16 * 32, 32, lane), o2);
        o3 = wmma_bf16(pf, frag_b(vtc + 3 * 16 * 32, 32, lane), o3);

        __syncthreads();   // all waves done reading buf[cur]; iter i+1 may overwrite it
    }

    __bf16* Ob = O + ((size_t)b * cT + qrow0) * cH + h * cHD;
#pragma unroll
    for (int r = 0; r < 8; ++r) {
        const int row = r + half * 8;
        const float inv = 1.0f / l_[r];
        Ob[(size_t)row * cH + 0  + col] = (__bf16)(o0[r] * inv);
        Ob[(size_t)row * cH + 16 + col] = (__bf16)(o1[r] * inv);
        Ob[(size_t)row * cH + 32 + col] = (__bf16)(o2[r] * inv);
        Ob[(size_t)row * cH + 48 + col] = (__bf16)(o3[r] * inv);
    }
}

// ---------------- host launcher ----------------
extern "C" void kernel_launch(void* const* d_in, const int* in_sizes, int n_in,
                              void* d_out, int out_size, void* d_ws, size_t ws_size,
                              hipStream_t stream) {
    const float* x  = (const float*)d_in[0];
    const float* wq = (const float*)d_in[1];
    const float* wk = (const float*)d_in[2];
    const float* wv = (const float*)d_in[3];
    const float* wo = (const float*)d_in[4];
    float* out = (float*)d_out;

    char* ws = (char*)d_ws;
    size_t off = 0;
    auto carve = [&](size_t elems) {
        __bf16* p = (__bf16*)(ws + off);
        off += ((elems * 2 + 255) & ~(size_t)255);   // keep 256B alignment
        return p;
    };
    const size_t nX  = (size_t)cB * cT * cH;            // 8.39M
    const size_t nWq = (size_t)cH * cH;                 // 4.19M
    const size_t nWk = (size_t)cNKV * cHD * cH;         // 1.05M
    const size_t nKV = (size_t)cB * cNKV * cT * cHD;    // 2.10M

    __bf16* xb  = carve(nX);
    __bf16* wqb = carve(nWq);
    __bf16* wkb = carve(nWk);
    __bf16* wvb = carve(nWk);
    __bf16* wob = carve(nWq);
    __bf16* Qb  = carve(nX);
    __bf16* Kb  = carve(nKV);
    __bf16* Vb  = carve(nKV);
    __bf16* Obf = carve(nX);

    // 1) convert to bf16
    mha_cvt_bf16<<<2048, 256, 0, stream>>>(x,  xb,  (int)nX);
    mha_cvt_bf16<<<2048, 256, 0, stream>>>(wq, wqb, (int)nWq);
    mha_cvt_bf16<<<1024, 256, 0, stream>>>(wk, wkb, (int)nWk);
    mha_cvt_bf16<<<1024, 256, 0, stream>>>(wv, wvb, (int)nWk);
    mha_cvt_bf16<<<2048, 256, 0, stream>>>(wo, wob, (int)nWq);

    const int M = cB * cT;          // 4096
    // 2) projections
    dim3 gq(cH / 64, M / 256);                 // Q/out proj: MI=4 -> 256 rows/block
    mha_gemm_bf16<0, 4><<<gq, 128, 0, stream>>>(xb, wqb, Qb, nullptr, M, cH, cH);
    dim3 gk((cNKV * cHD) / 64, M / 128);       // K,V: MI=2 -> 128 rows/block
    mha_gemm_bf16<1, 2><<<gk, 128, 0, stream>>>(xb, wkb, Kb, nullptr, M, cNKV * cHD, cH);
    mha_gemm_bf16<1, 2><<<gk, 128, 0, stream>>>(xb, wvb, Vb, nullptr, M, cNKV * cHD, cH);

    // 3) RoPE
    const int qpairs = cB * cT * cNH * (cHD / 2);   // 4,194,304
    const int kpairs = cB * cNKV * cT * (cHD / 2);  // 1,048,576
    mha_rope<<<(qpairs + 255) / 256, 256, 0, stream>>>(Qb, qpairs, (cHD / 2) * cNH);
    mha_rope<<<(kpairs + 255) / 256, 256, 0, stream>>>(Kb, kpairs, (cHD / 2));

    // 4) attention
    dim3 ga(cT / 64, cNH, cB);
    mha_attn<<<ga, 128, 0, stream>>>(Qb, Kb, Vb, Obf);

    // 5) output projection -> fp32 d_out
    mha_gemm_bf16<2, 4><<<gq, 128, 0, stream>>>(Obf, wob, nullptr, out, M, cH, cH);

    (void)in_sizes; (void)n_in; (void)out_size; (void)ws_size;
}
